// _VectorQuantizerBase_77781857731258
// MI455X (gfx1250) — matrix operand order repaired and checked by reading, not compiled, not run on hardware
//
#include <hip/hip_runtime.h>
#include <hip/hip_bf16.h>
#include <hip/hip_fp16.h>

typedef __attribute__((ext_vector_type(16))) _Float16 v16h;
typedef __attribute__((ext_vector_type(8)))  _Float16 v8h;
typedef __attribute__((ext_vector_type(8)))  float    v8f;

#define NUM_E 1024      // codebook entries (K)
#define DIM   256       // embedding dim (C)
#define NTOT  32768     // 32 * 32 * 32 pixels (N)
#define HW    1024      // 32*32 pixels per batch image
#define NBLK  32        // pixels handled per workgroup (2 x 16-wide sub-tiles)
#define LDSTR 34        // padded LDS row stride in floats (16*34 % 64 == 32 -> no bank conflict)

// ---------------------------------------------------------------------------
// Kernel 0: codebook -> f16 hi/lo split, ||e||^2, zero histogram counts
// ---------------------------------------------------------------------------
__global__ void vq_prep(const float* __restrict__ cb,
                        float* __restrict__ e_sq,
                        _Float16* __restrict__ cb_hi,
                        _Float16* __restrict__ cb_lo,
                        int* __restrict__ counts)
{
    int k = blockIdx.x * blockDim.x + threadIdx.x;   // 0..1023
    if (k >= NUM_E) return;
    counts[k] = 0;                                   // re-zero every launch
    const float* row = cb + (size_t)k * DIM;
    float s = 0.f;
    for (int c = 0; c < DIM; ++c) {
        float v = row[c];
        s += v * v;
        _Float16 h = (_Float16)v;
        cb_hi[(size_t)k * DIM + c] = h;
        cb_lo[(size_t)k * DIM + c] = (_Float16)(v - (float)h);
    }
    e_sq[k] = s;
}

// ---------------------------------------------------------------------------
// Kernel 1: fused distances GEMM (WMMA f16 hi/lo) + argmin + one-hot + hist
// Mapping: M = codebook index k, N = pixel n, K = channel c.
//   A[m,kk] = codebook[k0+m, c0+kk]   (row-major, contiguous)
//   B[kk,n] = z_e[b, c0+kk, p0+n]     (contiguous along n)
//   D[m,n]  = z.e dot tile -> dist = ||z||^2 + ||e||^2 - 2*D
// Each wave: 8 k-tiles x 2 n-sub-tiles; A fragments reused across both
// n-sub-tiles to halve codebook L2 traffic per output.
// ---------------------------------------------------------------------------
__global__ __launch_bounds__(256) void vq_main(
        const float*    __restrict__ z,       // [32,256,1024] fp32
        const _Float16* __restrict__ cb_hi,   // [1024,256]
        const _Float16* __restrict__ cb_lo,   // [1024,256]
        const float*    __restrict__ e_sq,    // [1024]
        float* __restrict__ dist_out,         // [32768,1024]
        float* __restrict__ idx_out,          // [32768] (as float)
        float* __restrict__ enc_out,          // [32768,1024]
        int*   __restrict__ counts)           // [1024]
{
    __shared__ float zs[DIM * LDSTR];   // z block [256c x 32n], padded rows
    __shared__ float zsq[NBLK];
    __shared__ float wmin[8][NBLK];
    __shared__ int   widx[8][NBLK];

    const int tid  = threadIdx.x;
    const int wave = tid >> 5;
    const int lane = tid & 31;
    const int hl   = lane >> 4;    // half-wave: 0 / 1
    const int nl   = lane & 15;    // D-tile column (pixel) / A row (code)

    const int n0 = blockIdx.x * NBLK;       // first pixel of this block
    const int b  = n0 / HW;                 // batch image
    const int p0 = n0 % HW;                 // pixel offset inside image

    // ---- stage z block [256c x 32n] into LDS (128B-row coalesced) ----
    {
        const int nn   = tid & 31;
        const int crow = tid >> 5;          // 0..7
        #pragma unroll
        for (int j = 0; j < 32; ++j) {
            int c = crow + j * 8;
            zs[c * LDSTR + nn] = z[((size_t)(b * DIM + c)) * HW + p0 + nn];
        }
    }
    __syncthreads();

    // ---- per-pixel squared norms ----
    if (tid < NBLK) {
        float s = 0.f;
        for (int c = 0; c < DIM; ++c) { float v = zs[c * LDSTR + tid]; s += v * v; }
        zsq[tid] = s;
    }
    __syncthreads();

    // ---- accumulators: 2 n-sub-tiles x 8 k-tiles per wave ----
    const int kw0 = wave * 128;
    v8f acc[2][8];
    {
        v8f zero = {};
        #pragma unroll
        for (int ns = 0; ns < 2; ++ns)
            #pragma unroll
            for (int t = 0; t < 8; ++t) acc[ns][t] = zero;
    }

    #pragma unroll
    for (int s = 0; s < 8; ++s) {             // K-loop: c0 = 32*s
        const int c0 = s * 32;

        // B fragments (z) for both 16-pixel sub-tiles: lane holds column
        // n = ns*16+nl; b[0..7] <-> K = 8*hl + 0..7 ; b[8..15] <-> K = 16+8*hl+0..7.
        v16h bh[2], bl[2];
        #pragma unroll
        for (int ns = 0; ns < 2; ++ns) {
            #pragma unroll
            for (int j = 0; j < 16; ++j) {
                float v = zs[(c0 + hl * 8 + (j & 7) + (j >> 3) * 16) * LDSTR + ns * 16 + nl];
                _Float16 h = (_Float16)v;
                bh[ns][j] = h;
                bl[ns][j] = (_Float16)(v - (float)h);
            }
        }

        #pragma unroll
        for (int t = 0; t < 8; ++t) {
            const int k = kw0 + t * 16 + nl;  // A row for this lane
            // A layout: a[0..7] <-> K = 8*hl + 0..7 ; a[8..15] <-> K = 16+8*hl+0..7
            const _Float16* ph = cb_hi + (size_t)k * DIM + c0 + hl * 8;
            const _Float16* pl = cb_lo + (size_t)k * DIM + c0 + hl * 8;
            v8h h0 = *(const v8h*)(ph);
            v8h h1 = *(const v8h*)(ph + 16);
            v8h l0 = *(const v8h*)(pl);
            v8h l1 = *(const v8h*)(pl + 16);
            v16h ah = __builtin_shufflevector(h0, h1, 0,1,2,3,4,5,6,7,8,9,10,11,12,13,14,15);
            v16h al = __builtin_shufflevector(l0, l1, 0,1,2,3,4,5,6,7,8,9,10,11,12,13,14,15);

            #pragma unroll
            for (int ns = 0; ns < 2; ++ns) {
                acc[ns][t] = __builtin_amdgcn_wmma_f32_16x16x32_f16(false, ah, false, bh[ns],
                                                                    (short)0, acc[ns][t], false, false);
                acc[ns][t] = __builtin_amdgcn_wmma_f32_16x16x32_f16(false, ah, false, bl[ns],
                                                                    (short)0, acc[ns][t], false, false);
                acc[ns][t] = __builtin_amdgcn_wmma_f32_16x16x32_f16(false, al, false, bh[ns],
                                                                    (short)0, acc[ns][t], false, false);
            }
        }
    }

    // ---- epilogue: distances, encodings zero-fill, local argmin ----
    #pragma unroll
    for (int ns = 0; ns < 2; ++ns) {
        float bestv = 3.4e38f;
        int   besti = 0;
        const int   np   = ns * 16 + nl;            // pixel within block
        const float zn   = zsq[np];
        const size_t nrow = (size_t)(n0 + np) * NUM_E;

        #pragma unroll
        for (int t = 0; t < 8; ++t) {
            const int kb = kw0 + t * 16 + 8 * hl;   // lane's 8 consecutive codes
            const float4 e0 = *(const float4*)(e_sq + kb);
            const float4 e1 = *(const float4*)(e_sq + kb + 4);
            float d[8];
            d[0] = zn + e0.x - 2.f * acc[ns][t][0];
            d[1] = zn + e0.y - 2.f * acc[ns][t][1];
            d[2] = zn + e0.z - 2.f * acc[ns][t][2];
            d[3] = zn + e0.w - 2.f * acc[ns][t][3];
            d[4] = zn + e1.x - 2.f * acc[ns][t][4];
            d[5] = zn + e1.y - 2.f * acc[ns][t][5];
            d[6] = zn + e1.z - 2.f * acc[ns][t][6];
            d[7] = zn + e1.w - 2.f * acc[ns][t][7];
            #pragma unroll
            for (int v = 0; v < 8; ++v) {
                if (d[v] < bestv) { bestv = d[v]; besti = kb + v; }
            }
            const size_t off = nrow + kb;
            *(float4*)(dist_out + off)     = make_float4(d[0], d[1], d[2], d[3]);
            *(float4*)(dist_out + off + 4) = make_float4(d[4], d[5], d[6], d[7]);
            const float4 z4 = make_float4(0.f, 0.f, 0.f, 0.f);
            *(float4*)(enc_out + off)     = z4;
            *(float4*)(enc_out + off + 4) = z4;
        }

        // combine the two half-wave lanes that share pixel np
        float ov = __shfl_xor(bestv, 16, 32);
        int   oi = __shfl_xor(besti, 16, 32);
        if (ov < bestv || (ov == bestv && oi < besti)) { bestv = ov; besti = oi; }
        if (hl == 0) { wmin[wave][np] = bestv; widx[wave][np] = besti; }
    }

    __threadfence();        // order encodings zero-stores before one-hot write
    __syncthreads();

    if (tid < NBLK) {
        float bv = wmin[0][tid]; int bi = widx[0][tid];
        #pragma unroll
        for (int w = 1; w < 8; ++w) {
            float v = wmin[w][tid]; int i = widx[w][tid];
            if (v < bv || (v == bv && i < bi)) { bv = v; bi = i; }
        }
        idx_out[n0 + tid] = (float)bi;
        enc_out[(size_t)(n0 + tid) * NUM_E + bi] = 1.0f;
        atomicAdd(&counts[bi], 1);
    }
}

// ---------------------------------------------------------------------------
// Kernel 2: perplexity from histogram
// ---------------------------------------------------------------------------
__global__ void vq_perp(const int* __restrict__ counts, float* __restrict__ perp_out)
{
    __shared__ float red[256];
    const int tid = threadIdx.x;
    float s = 0.f;
    #pragma unroll
    for (int j = 0; j < 4; ++j) {
        int k = tid + j * 256;
        float p = (float)counts[k] / (float)NTOT;
        s += p * logf(p + 1e-10f);
    }
    red[tid] = s;
    __syncthreads();
    for (int w = 128; w > 0; w >>= 1) {
        if (tid < w) red[tid] += red[tid + w];
        __syncthreads();
    }
    if (tid == 0) perp_out[0] = expf(-red[0]);
}

// ---------------------------------------------------------------------------
extern "C" void kernel_launch(void* const* d_in, const int* in_sizes, int n_in,
                              void* d_out, int out_size, void* d_ws, size_t ws_size,
                              hipStream_t stream)
{
    (void)in_sizes; (void)n_in; (void)out_size; (void)ws_size;

    const float* z_e = (const float*)d_in[0];   // [32,256,32,32] fp32
    const float* cb  = (const float*)d_in[1];   // [1024,256]    fp32

    // outputs concatenated flat in return order:
    // distances [N,K] | indices [N] | encodings [N,K] | perplexity [1]
    float* out      = (float*)d_out;
    float* dist_out = out;
    float* idx_out  = dist_out + (size_t)NTOT * NUM_E;
    float* enc_out  = idx_out + NTOT;
    float* perp_out = enc_out + (size_t)NTOT * NUM_E;

    // workspace layout
    char* ws = (char*)d_ws;
    int*      counts = (int*)ws;                               // 4 KB
    float*    e_sq   = (float*)(ws + 4096);                    // 4 KB
    _Float16* cb_hi  = (_Float16*)(ws + 8192);                 // 512 KB
    _Float16* cb_lo  = (_Float16*)(ws + 8192 + (size_t)NUM_E * DIM * 2);

    vq_prep<<<NUM_E / 256, 256, 0, stream>>>(cb, e_sq, cb_hi, cb_lo, counts);
    vq_main<<<NTOT / NBLK, 256, 0, stream>>>(z_e, cb_hi, cb_lo, e_sq,
                                             dist_out, idx_out, enc_out, counts);
    vq_perp<<<1, 256, 0, stream>>>(counts, perp_out);
}
